// ElecUpdate_32418413150494
// MI455X (gfx1250) — matrix operand order repaired and checked by reading, not compiled, not run on hardware
//
#include <hip/hip_runtime.h>

// ---------------------------------------------------------------------------
// Types
// ---------------------------------------------------------------------------
typedef __attribute__((ext_vector_type(4)))  float  v4f;
typedef __attribute__((ext_vector_type(8)))  float  v8f;
typedef __attribute__((ext_vector_type(8)))  __bf16 v8bf;
typedef __attribute__((ext_vector_type(16))) __bf16 v16bf;

#define N_ELEC 4096
#define J_NB   32
#define D_FEAT 256
#define LDS_STRIDE 272   // 256 + 16 bf16 pad -> spreads LDS banks

static __device__ __forceinline__ __bf16 f2bf(float f) {
    unsigned u = __builtin_bit_cast(unsigned, f);
    unsigned r = u + 0x7FFFu + ((u >> 16) & 1u);   // round-to-nearest-even
    unsigned short hs = (unsigned short)(r >> 16);
    return __builtin_bit_cast(__bf16, hs);
}

static __device__ __forceinline__ float silu(float x) {
    return x / (1.0f + __expf(-x));
}

// ---------------------------------------------------------------------------
// Kernel 1: transpose + convert weights to bf16.
// Wt[w][n*256 + k] = bf16( W_w[k*256 + n] )   -> B-fragment = contiguous 16 K's
// ---------------------------------------------------------------------------
__global__ void __launch_bounds__(256)
prep_kernel(const float* __restrict__ W1, const float* __restrict__ W2,
            const float* __restrict__ W3, __bf16* __restrict__ Wt) {
    int idx = blockIdx.x * 256 + threadIdx.x;      // 3 * 65536 total
    int w   = idx >> 16;
    int rem = idx & 65535;
    int n   = rem >> 8;
    int k   = rem & 255;
    const float* __restrict__ W = (w == 0) ? W1 : (w == 1) ? W2 : W3;
    Wt[idx] = f2bf(W[k * D_FEAT + n]);
}

// ---------------------------------------------------------------------------
// Kernel 2: msg[n,:] = sum_j gamma[n,j,:] * silu(h_nb[n,j,:] + h[n,:])
// One wave32 per row; lane owns 8 features. Spin select picks ONE of
// nb_same/nb_diff (halves streamed bytes). Non-temporal loads: single use.
// ---------------------------------------------------------------------------
__global__ void __launch_bounds__(256)
msg_kernel(const float* __restrict__ h, const float* __restrict__ gamma,
           const float* __restrict__ nb_same, const float* __restrict__ nb_diff,
           const int* __restrict__ spin, const int* __restrict__ spin_nb,
           float* __restrict__ msg) {
    const int wave = threadIdx.x >> 5;
    const int lane = threadIdx.x & 31;
    const int n    = blockIdx.x * 8 + wave;     // 512 blocks * 8 rows = 4096
    const int c    = lane * 8;                  // first feature for this lane

    const size_t rowD  = (size_t)n * D_FEAT + c;
    const size_t rowJD = (size_t)n * J_NB * D_FEAT + c;

    v4f h0 = *(const v4f*)(h + rowD);
    v4f h1 = *(const v4f*)(h + rowD + 4);
    const int sp = spin[n];

    v4f a0 = {0.f, 0.f, 0.f, 0.f};
    v4f a1 = {0.f, 0.f, 0.f, 0.f};

    const float* __restrict__ bs = nb_same + rowJD;
    const float* __restrict__ bd = nb_diff + rowJD;
    const float* __restrict__ bg = gamma   + rowJD;

    for (int j = 0; j < J_NB; ++j) {
        const float* src = (sp == spin_nb[n * J_NB + j]) ? bs : bd;
        v4f x0 = __builtin_nontemporal_load((const v4f*)(src + j * D_FEAT));
        v4f x1 = __builtin_nontemporal_load((const v4f*)(src + j * D_FEAT + 4));
        v4f g0 = __builtin_nontemporal_load((const v4f*)(bg  + j * D_FEAT));
        v4f g1 = __builtin_nontemporal_load((const v4f*)(bg  + j * D_FEAT + 4));
        #pragma unroll
        for (int e = 0; e < 4; ++e) {
            a0[e] += g0[e] * silu(x0[e] + h0[e]);
            a1[e] += g1[e] * silu(x1[e] + h1[e]);
        }
    }
    *(v4f*)(msg + rowD)     = a0;
    *(v4f*)(msg + rowD + 4) = a1;
}

// ---------------------------------------------------------------------------
// Kernel 3: fused 3-layer MLP with residuals, bf16 WMMA (f32 accumulate).
// Block = 16 rows x 256 cols; 8 waves, each owns two 16x16 column tiles.
// h1 fragments live in registers across layers for the final residual.
// ---------------------------------------------------------------------------
__global__ void __launch_bounds__(256)
mlp_kernel(const float* __restrict__ h, const float* __restrict__ msg,
           const __bf16* __restrict__ W1t, const float* __restrict__ b1,
           const __bf16* __restrict__ W2t, const float* __restrict__ b2,
           const __bf16* __restrict__ W3t, const float* __restrict__ b3,
           float* __restrict__ out) {
    __shared__ __align__(32) __bf16 Atile[16 * LDS_STRIDE];

    const int tid  = threadIdx.x;
    const int wave = tid >> 5;
    const int lane = tid & 31;
    const int m0   = blockIdx.x * 16;
    const int half = (lane < 16) ? 0 : 1;   // K-half select: k 16h..16h+15 of each 32-k step, same for A and B
    const int nloc = lane & 15;             // M for A-frag, N for B/C-frag
    const int n0a  = wave * 32;
    const int n0b  = wave * 32 + 16;

    // Stage h rows (16 x 256) into LDS as bf16 (vectorized: 64 v4f per row).
    for (int i = tid; i < 16 * 64; i += 256) {
        int r = i >> 6, cc = (i & 63) * 4;
        v4f v = *(const v4f*)(h + (size_t)(m0 + r) * D_FEAT + cc);
        __bf16* dst = &Atile[r * LDS_STRIDE + cc];
        dst[0] = f2bf(v[0]); dst[1] = f2bf(v[1]);
        dst[2] = f2bf(v[2]); dst[3] = f2bf(v[3]);
    }
    __syncthreads();

    const v8f vzero = {0.f, 0.f, 0.f, 0.f, 0.f, 0.f, 0.f, 0.f};

    // GEMM over K=256: A from LDS (contiguous 16 k per lane half), B from transposed bf16 W.
    auto gemm = [&](const __bf16* __restrict__ Wt, v8f& acc0, v8f& acc1) {
        acc0 = vzero;
        acc1 = vzero;
        #pragma unroll
        for (int k0 = 0; k0 < D_FEAT; k0 += 32) {
            const int kbase = k0 + half * 16;
            v8bf alo = *(const v8bf*)&Atile[nloc * LDS_STRIDE + kbase];
            v8bf ahi = *(const v8bf*)&Atile[nloc * LDS_STRIDE + kbase + 8];
            v16bf af = __builtin_shufflevector(alo, ahi,
                        0,1,2,3,4,5,6,7,8,9,10,11,12,13,14,15);
            v16bf bf0 = *(const v16bf*)&Wt[(size_t)(n0a + nloc) * D_FEAT + k0 + half * 16];
            v16bf bf1 = *(const v16bf*)&Wt[(size_t)(n0b + nloc) * D_FEAT + k0 + half * 16];
            acc0 = __builtin_amdgcn_wmma_f32_16x16x32_bf16(
                       false, af, false, bf0, (short)0, acc0, false, false);
            acc1 = __builtin_amdgcn_wmma_f32_16x16x32_bf16(
                       false, af, false, bf1, (short)0, acc1, false, false);
        }
    };

    v8f acc0, acc1;

    // ---- Layer 1: h1 = silu(h @ W1 + b1); t = h1 + msg -------------------
    gemm(W1t, acc0, acc1);
    const float bia0 = b1[n0a + nloc];
    const float bia1 = b1[n0b + nloc];
    v8f h1f0, h1f1, t0, t1;
    #pragma unroll
    for (int r = 0; r < 8; ++r) {
        const int M = r + half * 8;
        float va = silu(acc0[r] + bia0);
        float vb = silu(acc1[r] + bia1);
        h1f0[r] = va;
        h1f1[r] = vb;
        t0[r] = va + msg[(size_t)(m0 + M) * D_FEAT + n0a + nloc];
        t1[r] = vb + msg[(size_t)(m0 + M) * D_FEAT + n0b + nloc];
    }
    __syncthreads();                       // all waves done reading h tile
    #pragma unroll
    for (int r = 0; r < 8; ++r) {
        const int M = r + half * 8;
        Atile[M * LDS_STRIDE + n0a + nloc] = f2bf(t0[r]);
        Atile[M * LDS_STRIDE + n0b + nloc] = f2bf(t1[r]);
    }
    __syncthreads();

    // ---- Layer 2: o2 = silu(t @ W2 + b2) ---------------------------------
    gemm(W2t, acc0, acc1);
    const float b2a = b2[n0a + nloc];
    const float b2b = b2[n0b + nloc];
    v8f o2a, o2b;
    #pragma unroll
    for (int r = 0; r < 8; ++r) {
        o2a[r] = silu(acc0[r] + b2a);
        o2b[r] = silu(acc1[r] + b2b);
    }
    __syncthreads();                       // all waves done reading t tile
    #pragma unroll
    for (int r = 0; r < 8; ++r) {
        const int M = r + half * 8;
        Atile[M * LDS_STRIDE + n0a + nloc] = f2bf(o2a[r]);
        Atile[M * LDS_STRIDE + n0b + nloc] = f2bf(o2b[r]);
    }
    __syncthreads();

    // ---- Layer 3: out = silu(o2 @ W3 + b3) + h1 --------------------------
    gemm(W3t, acc0, acc1);
    const float b3a = b3[n0a + nloc];
    const float b3b = b3[n0b + nloc];
    #pragma unroll
    for (int r = 0; r < 8; ++r) {
        const int M = r + half * 8;
        out[(size_t)(m0 + M) * D_FEAT + n0a + nloc] = silu(acc0[r] + b3a) + h1f0[r];
        out[(size_t)(m0 + M) * D_FEAT + n0b + nloc] = silu(acc1[r] + b3b) + h1f1[r];
    }
}

// ---------------------------------------------------------------------------
// Launch
// ---------------------------------------------------------------------------
extern "C" void kernel_launch(void* const* d_in, const int* in_sizes, int n_in,
                              void* d_out, int out_size, void* d_ws, size_t ws_size,
                              hipStream_t stream) {
    const float* h      = (const float*)d_in[0];
    const float* gamma  = (const float*)d_in[1];
    const float* nbs    = (const float*)d_in[2];
    const float* nbd    = (const float*)d_in[3];
    const int*   spin   = (const int*)d_in[4];
    const int*   spinnb = (const int*)d_in[5];
    const float* W1     = (const float*)d_in[6];
    const float* b1     = (const float*)d_in[7];
    const float* W2     = (const float*)d_in[8];
    const float* b2     = (const float*)d_in[9];
    const float* W3     = (const float*)d_in[10];
    const float* b3     = (const float*)d_in[11];
    float* out = (float*)d_out;

    // Workspace layout: [ msg: 4096*256 f32 = 4 MB ][ Wt: 3*65536 bf16 = 384 KB ]
    float*  msg = (float*)d_ws;
    __bf16* Wt  = (__bf16*)((char*)d_ws + (size_t)N_ELEC * D_FEAT * sizeof(float));

    prep_kernel<<<768, 256, 0, stream>>>(W1, W2, W3, Wt);
    msg_kernel<<<N_ELEC / 8, 256, 0, stream>>>(h, gamma, nbs, nbd, spin, spinnb, msg);
    mlp_kernel<<<N_ELEC / 16, 256, 0, stream>>>(h, msg,
                                                Wt,             b1,
                                                Wt + 65536,     b2,
                                                Wt + 2 * 65536, b3,
                                                out);
}
